// AFW_63118839382657
// MI455X (gfx1250) — matrix operand order repaired and checked by reading, not compiled, not run on hardware
//
#include <hip/hip_runtime.h>
#include <hip/hip_bf16.h>

#define LAYERS 2
#define NMODS  3
#define DIM    512
#define RNK    32
#define BSZ    16
#define TLEN   512
#define BETAC  0.1f

typedef _Float16 h16;
typedef __attribute__((ext_vector_type(16))) _Float16 v16h;
typedef __attribute__((ext_vector_type(8)))  _Float16 v8h;
typedef __attribute__((ext_vector_type(8)))  float    v8f;

// K index held by half h (0..15) of the 16-bit A-fragment, for lane-half hi (0/1).
// Per ISA 7.12.2: lanes 0-15 hold K {0..7,16..23}, lanes 16-31 hold K {8..15,24..31}.
__device__ __forceinline__ int a_kpat(int h, int hi) {
  return h + ((h >= 8) ? 8 : 0) + hi * 8;
}

// ---------------------------------------------------------------- prep kernels
__global__ void k_copyx(const float* __restrict__ xa, const float* __restrict__ xt,
                        const float* __restrict__ xv, float* __restrict__ Xcur,
                        h16* __restrict__ Xh) {
  const size_t n1 = (size_t)BSZ * TLEN * DIM;
  int m = blockIdx.y;
  const float* src = (m == 0) ? xa : (m == 1) ? xt : xv;
  size_t i = (size_t)blockIdx.x * blockDim.x + threadIdx.x;
  float v = src[i];
  Xcur[(size_t)m * n1 + i] = v;
  Xh[(size_t)m * n1 + i] = (h16)v;
}

// [L*3][D][R] f32 -> [L*3][R][D] f16  (proj weights, transposed for contiguous B-frags)
__global__ void k_cvt_wT(const float* __restrict__ src, h16* __restrict__ dst) {
  size_t i = (size_t)blockIdx.x * blockDim.x + threadIdx.x;   // over L*3*D*R
  size_t lm = i / (DIM * RNK);
  size_t rem = i % (DIM * RNK);
  size_t k = rem / RNK, r = rem % RNK;
  dst[lm * (DIM * RNK) + r * DIM + k] = (h16)src[i];
}

// [L*3][1024][D] f32 -> [L*3][D][1024] f16  (Watt, transposed)
__global__ void k_cvt_wattT(const float* __restrict__ src, h16* __restrict__ dst) {
  size_t i = (size_t)blockIdx.x * blockDim.x + threadIdx.x;   // over L*3*1024*D
  size_t lm = i / ((size_t)RNK * RNK * DIM);
  size_t rem = i % ((size_t)RNK * RNK * DIM);
  size_t rr = rem / DIM, d = rem % DIM;
  dst[lm * ((size_t)RNK * RNK * DIM) + d * (RNK * RNK) + rr] = (h16)src[i];
}

// ------------------------------------------------- stage A: u,v projections
// grid (TLEN/64, 3*B), block 256 (8 waves: 4 t-tiles x 2 r-halves)
__global__ void __launch_bounds__(256) k_proj_uv(
    const h16* __restrict__ Xh,
    const h16* __restrict__ Wq1T, const h16* __restrict__ Wq2T,
    const h16* __restrict__ Wk1T, const h16* __restrict__ Wk2T,   // [3][R][D] (layer slice)
    const float* __restrict__ bq1, const float* __restrict__ bq2,
    const float* __restrict__ bk1, const float* __restrict__ bk2, // [3][R] (layer slice)
    float* __restrict__ U, float* __restrict__ V) {               // [3*B][T][R]
  int mb = blockIdx.y;
  int m = mb >> 4;
  int wid = threadIdx.x >> 5, lane = threadIdx.x & 31;
  int lane15 = lane & 15, hi = lane >> 4;
  int t0 = blockIdx.x * 64 + (wid >> 1) * 16;
  int r0 = (wid & 1) * 16;

  const h16* xrow = Xh + ((size_t)mb * TLEN + (t0 + lane15)) * DIM;
  size_t woff = (size_t)m * RNK * DIM + (size_t)(r0 + lane15) * DIM;
  const h16* w1 = Wq1T + woff;
  const h16* w2 = Wq2T + woff;
  const h16* w3 = Wk1T + woff;
  const h16* w4 = Wk2T + woff;

  v8f c1 = {}, c2 = {}, c3 = {}, c4 = {};
  for (int kc = 0; kc < DIM / 32; ++kc) {
    int k0 = kc * 32;
    v16h a;
    {
      v8h s0 = *(const v8h*)(xrow + k0 + hi * 16);
      v8h s1 = *(const v8h*)(xrow + k0 + hi * 16 + 8);
#pragma unroll
      for (int h = 0; h < 8; ++h) { a[h] = s0[h]; a[h + 8] = s1[h]; }
    }
    int kb = k0 + hi * 16;
    v16h b1, b2, b3, b4;
    {
      v8h p0 = *(const v8h*)(w1 + kb); v8h p1 = *(const v8h*)(w1 + kb + 8);
      v8h q0 = *(const v8h*)(w2 + kb); v8h q1 = *(const v8h*)(w2 + kb + 8);
      v8h r0v = *(const v8h*)(w3 + kb); v8h r1v = *(const v8h*)(w3 + kb + 8);
      v8h s0 = *(const v8h*)(w4 + kb); v8h s1 = *(const v8h*)(w4 + kb + 8);
#pragma unroll
      for (int h = 0; h < 8; ++h) {
        b1[h] = p0[h]; b1[h + 8] = p1[h];
        b2[h] = q0[h]; b2[h + 8] = q1[h];
        b3[h] = r0v[h]; b3[h + 8] = r1v[h];
        b4[h] = s0[h]; b4[h + 8] = s1[h];
      }
    }
    c1 = __builtin_amdgcn_wmma_f32_16x16x32_f16(false, a, false, b1, (short)0, c1, false, false);
    c2 = __builtin_amdgcn_wmma_f32_16x16x32_f16(false, a, false, b2, (short)0, c2, false, false);
    c3 = __builtin_amdgcn_wmma_f32_16x16x32_f16(false, a, false, b3, (short)0, c3, false, false);
    c4 = __builtin_amdgcn_wmma_f32_16x16x32_f16(false, a, false, b4, (short)0, c4, false, false);
  }
  int r = r0 + lane15;
  float vb1 = bq1[m * RNK + r], vb2 = bq2[m * RNK + r];
  float vb3 = bk1[m * RNK + r], vb4 = bk2[m * RNK + r];
#pragma unroll
  for (int i = 0; i < 8; ++i) {
    int t = t0 + i + 8 * hi;
    size_t o = ((size_t)mb * TLEN + t) * RNK + r;
    U[o] = (c1[i] + vb1) * (c3[i] + vb3);
    V[o] = (c2[i] + vb2) * (c4[i] + vb4);
  }
}

// ------------------------------------------------- stage B: Mqk = (1/T) u^T v
// grid 3*B blocks, 128 threads (4 waves = 2x2 16x16 tiles of the 32x32 result)
__global__ void __launch_bounds__(128) k_mqk(
    const float* __restrict__ U, const float* __restrict__ V,
    float* __restrict__ Mqk) {
  int mb = blockIdx.x;
  int wid = threadIdx.x >> 5, lane = threadIdx.x & 31;
  int lane15 = lane & 15, hi = lane >> 4;
  int r1_0 = (wid >> 1) * 16, r2_0 = (wid & 1) * 16;
  const float* ub = U + (size_t)mb * TLEN * RNK;
  const float* vb = V + (size_t)mb * TLEN * RNK;
  v8f c = {};
  for (int tc = 0; tc < TLEN / 32; ++tc) {
    int t0 = tc * 32;
    v16h a, b;
#pragma unroll
    for (int h = 0; h < 16; ++h) {
      a[h] = (h16)ub[(size_t)(t0 + a_kpat(h, hi)) * RNK + (r1_0 + lane15)];
      b[h] = (h16)vb[(size_t)(t0 + a_kpat(h, hi)) * RNK + (r2_0 + lane15)];
    }
    c = __builtin_amdgcn_wmma_f32_16x16x32_f16(false, a, false, b, (short)0, c, false, false);
  }
  const float scale = 1.0f / (float)TLEN;
#pragma unroll
  for (int i = 0; i < 8; ++i) {
    int r1 = r1_0 + i + 8 * hi;
    Mqk[(size_t)mb * (RNK * RNK) + r1 * RNK + (r2_0 + lane15)] = c[i] * scale;
  }
}

// ------------------------------------------------- stage C: P_j = prod Mqk (fp32)
__global__ void __launch_bounds__(256) k_pmat(const float* __restrict__ Mqk,
                                              float* __restrict__ P) {
  int jb = blockIdx.x;            // j*16+b
  int j = jb >> 4, b = jb & 15;
  int l0 = (j == 0) ? 1 : 0;
  int l1 = (j == 2) ? 1 : 2;
  __shared__ float As[RNK * RNK], Bs[RNK * RNK];
  int tid = threadIdx.x;
#pragma unroll
  for (int q = 0; q < 4; ++q) {
    int idx = tid + q * 256;
    As[idx] = Mqk[(size_t)(l0 * 16 + b) * (RNK * RNK) + idx];
    Bs[idx] = Mqk[(size_t)(l1 * 16 + b) * (RNK * RNK) + idx];
  }
  __syncthreads();
  int row = tid >> 3, cg = (tid & 7) * 4;
  float acc0 = 0.f, acc1 = 0.f, acc2 = 0.f, acc3 = 0.f;
  for (int k = 0; k < RNK; ++k) {
    float av = As[row * RNK + k];
    acc0 += av * Bs[k * RNK + cg + 0];
    acc1 += av * Bs[k * RNK + cg + 1];
    acc2 += av * Bs[k * RNK + cg + 2];
    acc3 += av * Bs[k * RNK + cg + 3];
  }
  size_t o = (size_t)jb * (RNK * RNK) + row * RNK + cg;
  P[o + 0] = acc0; P[o + 1] = acc1; P[o + 2] = acc2; P[o + 3] = acc3;
}

// ------------------------------------------------- stage C2: w = v @ P  (K=32, 1 WMMA)
// grid (TLEN/64, 3*B), block 256
__global__ void __launch_bounds__(256) k_wmat(
    const float* __restrict__ V, const float* __restrict__ P,
    float* __restrict__ Wout) {
  int jb = blockIdx.y;
  int wid = threadIdx.x >> 5, lane = threadIdx.x & 31;
  int lane15 = lane & 15, hi = lane >> 4;
  int t0 = blockIdx.x * 64 + (wid >> 1) * 16;
  int r2_0 = (wid & 1) * 16;
  const float* vrow = V + ((size_t)jb * TLEN + (t0 + lane15)) * RNK;
  const float* pb = P + (size_t)jb * (RNK * RNK);
  v16h a, b;
#pragma unroll
  for (int h = 0; h < 16; ++h) {
    a[h] = (h16)vrow[a_kpat(h, hi)];
    b[h] = (h16)pb[a_kpat(h, hi) * RNK + (r2_0 + lane15)];
  }
  v8f c = {};
  c = __builtin_amdgcn_wmma_f32_16x16x32_f16(false, a, false, b, (short)0, c, false, false);
#pragma unroll
  for (int i = 0; i < 8; ++i) {
    int t = t0 + i + 8 * hi;
    Wout[((size_t)jb * TLEN + t) * RNK + (r2_0 + lane15)] = c[i];
  }
}

// ------------------------------------------------- stage D: out = ((u⊗w)·Watt + batt + β) ⊙ X
// grid (TLEN/32, DIM/256, 3*B), block 256 (8 waves: 2 t-tiles x 4 d-strips of 64)
__global__ void __launch_bounds__(256) k_bigproj(
    const float* __restrict__ U, const float* __restrict__ Wm,
    const h16* __restrict__ WattT,   // [3][D][1024] (layer slice)
    const float* __restrict__ batt,  // [3][D] (layer slice)
    const float* Xin, float* Xout, h16* XhOut) {
  int jb = blockIdx.z;
  int j = jb >> 4;
  int wid = threadIdx.x >> 5, lane = threadIdx.x & 31;
  int lane15 = lane & 15, hi = lane >> 4;
  int tb = blockIdx.x * 32;
  int t0 = tb + (wid >> 2) * 16;
  int d0 = blockIdx.y * 256 + (wid & 3) * 64;

  __shared__ float u_s[32][RNK];
  __shared__ float w_s[32][RNK];
  {
    int tid = threadIdx.x;
#pragma unroll
    for (int q = 0; q < 4; ++q) {
      int idx = tid * 4 + q;           // 0..1023
      int row = idx >> 5, col = idx & 31;
      u_s[row][col] = U[((size_t)jb * TLEN + tb + row) * RNK + col];
      w_s[row][col] = Wm[((size_t)jb * TLEN + tb + row) * RNK + col];
    }
  }
  __syncthreads();

  int tl = (wid >> 2) * 16 + lane15;
  const h16* wattBase = WattT + (size_t)j * DIM * (RNK * RNK);

  // Hoist the w-row fragment (A-fragment lane permutation) out of the K loop:
  // it is invariant across kc, so each chunk's A-fragment is a packed-f16 splat-mul.
  v16h wf;
#pragma unroll
  for (int h = 0; h < 16; ++h) wf[h] = (h16)w_s[tl][h + 16 * hi];

  v8f c0 = {}, c1 = {}, c2 = {}, c3 = {};
  for (int kc = 0; kc < RNK; ++kc) {   // kc == r1; K-chunk is rank-1: u[t,r1]*w[t,:]
    h16 uh = (h16)u_s[tl][kc];
    v16h us;
#pragma unroll
    for (int h = 0; h < 16; ++h) us[h] = uh;
    v16h a = us * wf;                  // 8x v_pk_mul_f16
    size_t rrBase = (size_t)kc * 32 + hi * 16;
    v16h bf0, bf1, bf2, bf3;
#pragma unroll
    for (int dd = 0; dd < 4; ++dd) {
      const h16* bp = wattBase + (size_t)(d0 + dd * 16 + lane15) * (RNK * RNK) + rrBase;
      v8h s0 = *(const v8h*)(bp);
      v8h s1 = *(const v8h*)(bp + 8);
      v16h bf;
#pragma unroll
      for (int h = 0; h < 8; ++h) { bf[h] = s0[h]; bf[h + 8] = s1[h]; }
      if (dd == 0) bf0 = bf; else if (dd == 1) bf1 = bf;
      else if (dd == 2) bf2 = bf; else bf3 = bf;
    }
    c0 = __builtin_amdgcn_wmma_f32_16x16x32_f16(false, a, false, bf0, (short)0, c0, false, false);
    c1 = __builtin_amdgcn_wmma_f32_16x16x32_f16(false, a, false, bf1, (short)0, c1, false, false);
    c2 = __builtin_amdgcn_wmma_f32_16x16x32_f16(false, a, false, bf2, (short)0, c2, false, false);
    c3 = __builtin_amdgcn_wmma_f32_16x16x32_f16(false, a, false, bf3, (short)0, c3, false, false);
  }
#pragma unroll
  for (int dd = 0; dd < 4; ++dd) {
    v8f c = (dd == 0) ? c0 : (dd == 1) ? c1 : (dd == 2) ? c2 : c3;
    int d = d0 + dd * 16 + lane15;
    float bv = batt[(size_t)j * DIM + d] + BETAC;
#pragma unroll
    for (int i = 0; i < 8; ++i) {
      int t = t0 + i + 8 * hi;
      size_t o = ((size_t)jb * TLEN + t) * DIM + d;
      float val = (c[i] + bv) * Xin[o];
      Xout[o] = val;
      if (XhOut) XhOut[o] = (h16)val;
    }
  }
}

// ---------------------------------------------------------------- launcher
extern "C" void kernel_launch(void* const* d_in, const int* in_sizes, int n_in,
                              void* d_out, int out_size, void* d_ws, size_t ws_size,
                              hipStream_t stream) {
  (void)in_sizes; (void)n_in; (void)out_size; (void)ws_size;
  const float* xa   = (const float*)d_in[0];
  const float* xt   = (const float*)d_in[1];
  const float* xv   = (const float*)d_in[2];
  const float* Wq1  = (const float*)d_in[3];
  const float* bq1  = (const float*)d_in[4];
  const float* Wq2  = (const float*)d_in[5];
  const float* bq2  = (const float*)d_in[6];
  const float* Wk1  = (const float*)d_in[7];
  const float* bk1  = (const float*)d_in[8];
  const float* Wk2  = (const float*)d_in[9];
  const float* bk2  = (const float*)d_in[10];
  const float* Watt = (const float*)d_in[11];
  const float* batt = (const float*)d_in[12];

  char* ws = (char*)d_ws;
  size_t off = 0;
  auto wsalloc = [&](size_t bytes) -> void* {
    void* p = (void*)(ws + off);
    off += (bytes + 255) & ~(size_t)255;
    return p;
  };
  const size_t NX = (size_t)NMODS * BSZ * TLEN * DIM;
  const size_t SW = (size_t)LAYERS * NMODS * DIM * RNK;             // per weight class
  const size_t SA = (size_t)LAYERS * NMODS * RNK * RNK * DIM;
  const size_t NU = (size_t)NMODS * BSZ * TLEN * RNK;
  float* Xcur  = (float*)wsalloc(NX * 4);
  h16*   Xh    = (h16*)wsalloc(NX * 2);
  h16*   WhT   = (h16*)wsalloc(4 * SW * 2);
  h16*   WattT = (h16*)wsalloc(SA * 2);
  float* Ub    = (float*)wsalloc(NU * 4);
  float* Vb    = (float*)wsalloc(NU * 4);
  float* Wb    = (float*)wsalloc(NU * 4);
  float* Mqkb  = (float*)wsalloc((size_t)NMODS * BSZ * RNK * RNK * 4);
  float* Pb    = (float*)wsalloc((size_t)NMODS * BSZ * RNK * RNK * 4);

  // prep: stage X (f32 + f16 copies), transposed f16 weights
  k_copyx<<<dim3((BSZ * TLEN * DIM) / 256, 3), 256, 0, stream>>>(xa, xt, xv, Xcur, Xh);
  k_cvt_wT<<<(unsigned)(SW / 256), 256, 0, stream>>>(Wq1, WhT + 0 * SW);
  k_cvt_wT<<<(unsigned)(SW / 256), 256, 0, stream>>>(Wq2, WhT + 1 * SW);
  k_cvt_wT<<<(unsigned)(SW / 256), 256, 0, stream>>>(Wk1, WhT + 2 * SW);
  k_cvt_wT<<<(unsigned)(SW / 256), 256, 0, stream>>>(Wk2, WhT + 3 * SW);
  k_cvt_wattT<<<(unsigned)(SA / 256), 256, 0, stream>>>(Watt, WattT);

  for (int li = 0; li < LAYERS; ++li) {
    size_t wlo = (size_t)li * NMODS * DIM * RNK;
    size_t blo = (size_t)li * NMODS * RNK;
    k_proj_uv<<<dim3(TLEN / 64, NMODS * BSZ), 256, 0, stream>>>(
        Xh, WhT + 0 * SW + wlo, WhT + 1 * SW + wlo, WhT + 2 * SW + wlo, WhT + 3 * SW + wlo,
        bq1 + blo, bq2 + blo, bk1 + blo, bk2 + blo, Ub, Vb);
    k_mqk<<<NMODS * BSZ, 128, 0, stream>>>(Ub, Vb, Mqkb);
    k_pmat<<<NMODS * BSZ, 256, 0, stream>>>(Mqkb, Pb);
    k_wmat<<<dim3(TLEN / 64, NMODS * BSZ), 256, 0, stream>>>(Vb, Pb, Wb);
    const bool last = (li == LAYERS - 1);
    k_bigproj<<<dim3(TLEN / 32, DIM / 256, NMODS * BSZ), 256, 0, stream>>>(
        Ub, Wb, WattT + (size_t)li * NMODS * RNK * RNK * DIM,
        batt + (size_t)li * NMODS * DIM, Xcur,
        last ? (float*)d_out : Xcur,
        last ? (h16*)nullptr : Xh);
  }
}
